// IpaAtom10DenoiserV3_2430951489749
// MI455X (gfx1250) — matrix operand order repaired and checked by reading, not compiled, not run on hardware
//
#include <hip/hip_runtime.h>
#include <hip/hip_bf16.h>

// ---- problem constants (match reference) ----
#define BATCH  2
#define NATOM  10240
#define NRESC  1024
#define APR    10
#define WQ     32
#define HKEY   128
#define HEADS  8
#define CATOM  256
#define CS     384
#define KBLK   320          // NATOM / WQ
#define DH     32           // CATOM / HEADS
#define MROWS  (BATCH*NATOM) // 20480
#define LN_EPS 1e-5f

typedef __bf16 v16bf __attribute__((ext_vector_type(16)));
typedef float  v8f   __attribute__((ext_vector_type(8)));

struct alignas(16) U4 { unsigned x, y, z, w; };

union AFrag {
  v16bf v;
  unsigned u[8];
  unsigned short h[16];
};

__device__ __forceinline__ v8f zero8() {
  v8f z; for (int i = 0; i < 8; ++i) z[i] = 0.0f; return z;
}

__device__ __forceinline__ unsigned short f2bf(float f) {
  unsigned u = __float_as_uint(f);
  u += 0x7fffu + ((u >> 16) & 1u);   // round-to-nearest-even
  return (unsigned short)(u >> 16);
}

// A-matrix 16x32 bf16 fragment: lane holds row M=lane%16,
// K elems {koff..koff+7} and {koff+16..koff+23}, koff = (lane>=16)*8.
__device__ __forceinline__ void loadA(AFrag& f, const unsigned short* p) {
  *(U4*)&f.u[0] = *(const U4*)p;
  *(U4*)&f.u[4] = *(const U4*)(p + 16);
}
// Row load: 16 contiguous bf16 p[0..15] -> elems 0..15 (used to read V rows
// for the LDS transpose; WMMA B-frags here are loaded with loadA).
__device__ __forceinline__ void loadB(AFrag& f, const unsigned short* p) {
  *(U4*)&f.u[0] = *(const U4*)p;
  *(U4*)&f.u[4] = *(const U4*)(p + 8);
}

__device__ __forceinline__ v8f wmma_bf16(const AFrag& a, const AFrag& b, v8f c) {
  return __builtin_amdgcn_wmma_f32_16x16x32_bf16(
      /*neg_a=*/false, a.v, /*neg_b=*/false, b.v,
      /*c_mod=*/(short)0, c, /*reuse_a=*/false, /*reuse_b=*/false);
}

// ---------------------------------------------------------------------------
// f32 -> bf16 copy / transpose (weight prep so B-frags are row-contiguous)
// ---------------------------------------------------------------------------
__global__ void k_cvt_bf16(const float* __restrict__ src,
                           unsigned short* __restrict__ dst, int n) {
  int i = blockIdx.x * blockDim.x + threadIdx.x;
  int st = gridDim.x * blockDim.x;
  for (; i < n; i += st) dst[i] = f2bf(src[i]);
}

// src [rows, cols] f32 -> dst [cols, rows] bf16
__global__ void k_transpose_bf16(const float* __restrict__ src,
                                 unsigned short* __restrict__ dst,
                                 int rows, int cols) {
  int i = blockIdx.x * blockDim.x + threadIdx.x;
  int n = rows * cols;
  int st = gridDim.x * blockDim.x;
  for (; i < n; i += st) {
    int r = i / cols, c = i % cols;
    dst[c * rows + r] = f2bf(src[i]);
  }
}

// ---------------------------------------------------------------------------
// GEMM: D[M,256] = A[M,256] x W  (W given transposed: BT[256out,256in] bf16)
// block = 8 waves = 128x64 tile, wave = 32x32 tile (2x2 WMMA), k-steps of 32
// ---------------------------------------------------------------------------
__global__ __launch_bounds__(256)
void k_gemm_qkv(const unsigned short* __restrict__ A,
                const unsigned short* __restrict__ BT,
                unsigned short* __restrict__ D) {
  const int lane = threadIdx.x & 31, wave = threadIdx.x >> 5;
  const int lr = lane & 15, hi = lane >> 4;
  const int wm = wave & 3, wn = wave >> 2;
  const int rowBase = blockIdx.x * 128 + wm * 32;
  const int colBase = blockIdx.y * 64 + wn * 32;
  v8f acc[2][2];
  for (int mt = 0; mt < 2; ++mt) for (int nt = 0; nt < 2; ++nt) acc[mt][nt] = zero8();

  for (int k0 = 0; k0 < CATOM; k0 += 32) {
    AFrag a[2], b[2];
    for (int mt = 0; mt < 2; ++mt)
      loadA(a[mt], A + (size_t)(rowBase + mt * 16 + lr) * CATOM + k0 + hi * 8);
    for (int nt = 0; nt < 2; ++nt)
      loadA(b[nt], BT + (size_t)(colBase + nt * 16 + lr) * CATOM + k0 + hi * 8);
    if (k0 + 32 < CATOM)
      __builtin_prefetch(A + (size_t)(rowBase + lr) * CATOM + k0 + 32, 0, 3);
    for (int mt = 0; mt < 2; ++mt)
      for (int nt = 0; nt < 2; ++nt)
        acc[mt][nt] = wmma_bf16(a[mt], b[nt], acc[mt][nt]);
  }
  // C layout: VGPR j holds (M = j + 8*hi, N = lr)
  for (int mt = 0; mt < 2; ++mt)
    for (int nt = 0; nt < 2; ++nt)
      for (int j = 0; j < 8; ++j) {
        int r = rowBase + mt * 16 + j + hi * 8;
        int c = colBase + nt * 16 + lr;
        D[(size_t)r * CATOM + c] = f2bf(acc[mt][nt][j]);
      }
}

// ---------------------------------------------------------------------------
// Local windowed attention. grid = (KBLK, BATCH), 8 waves, wave == head.
// Window for block kb = atoms [32*kb-48, 32*kb+80) (bounds-masked) — this is
// exactly the single_to_keys half-block gather, computed implicitly.
// Dynamic LDS per head: probs[32][136] bf16 + vT[32][136] bf16 (row pad 8).
// ---------------------------------------------------------------------------
__global__ __launch_bounds__(256)
void k_attention(const unsigned short* __restrict__ Q,
                 const unsigned short* __restrict__ Kb,
                 const unsigned short* __restrict__ Vb,
                 const float* __restrict__ pair_bias,
                 const float* __restrict__ atom_mask,
                 unsigned short* __restrict__ AO) {
  extern __shared__ unsigned short smem[];
  const int lane = threadIdx.x & 31, h = threadIdx.x >> 5;
  const int lr = lane & 15, hi = lane >> 4;
  const int kb = blockIdx.x, b = blockIdx.y;
  const int q0 = kb * WQ;
  const int kstart = kb * WQ - 48;
  unsigned short* probs = smem + h * (2 * 32 * 136);
  unsigned short* vT    = probs + 32 * 136;

  AFrag qa[2];
  for (int mt = 0; mt < 2; ++mt)
    loadA(qa[mt], Q + (size_t)(b * NATOM + q0 + mt * 16 + lr) * CATOM + h * DH + hi * 8);

  // scores [32 x 128]: 2x8 WMMA tiles, K = dh = 32 (single WMMA each)
  v8f sc[2][8];
  for (int nt = 0; nt < 8; ++nt) {
    AFrag kf;
    int key = kstart + nt * 16 + lr;
    if (key >= 0 && key < NATOM)
      loadA(kf, Kb + (size_t)(b * NATOM + key) * CATOM + h * DH + hi * 8);
    else
      for (int i = 0; i < 8; ++i) kf.u[i] = 0u;
    for (int mt = 0; mt < 2; ++mt)
      sc[mt][nt] = wmma_bf16(qa[mt], kf, zero8());
  }

  // scale + pair_bias + mask
  const float scale = 0.17677669529663687f;  // 1/sqrt(32)
  float kmv[8];
  for (int nt = 0; nt < 8; ++nt) {
    int key = kstart + nt * 16 + lr;
    kmv[nt] = (key >= 0 && key < NATOM) ? atom_mask[b * NATOM + key] : 0.0f;
  }
  for (int mt = 0; mt < 2; ++mt)
    for (int j = 0; j < 8; ++j) {
      int w = mt * 16 + j + hi * 8;
      float qm = atom_mask[b * NATOM + q0 + w];
      for (int nt = 0; nt < 8; ++nt) {
        size_t bi = ((((size_t)(b * KBLK + kb)) * WQ + w) * HKEY + nt * 16 + lr) * HEADS + h;
        float v = sc[mt][nt][j] * scale + pair_bias[bi];
        sc[mt][nt][j] = (qm > 0.0f && kmv[nt] > 0.0f) ? v : -1e9f;
      }
    }

  // softmax over 128 keys: row lives in one 16-lane half x 8 nt-regs
  for (int mt = 0; mt < 2; ++mt)
    for (int j = 0; j < 8; ++j) {
      float m = -3.4e38f;
      for (int nt = 0; nt < 8; ++nt) m = fmaxf(m, sc[mt][nt][j]);
      for (int o = 1; o < 16; o <<= 1) m = fmaxf(m, __shfl_xor(m, o, 16));
      float s = 0.0f;
      for (int nt = 0; nt < 8; ++nt) {
        float e = __expf(sc[mt][nt][j] - m);
        sc[mt][nt][j] = e; s += e;
      }
      for (int o = 1; o < 16; o <<= 1) s += __shfl_xor(s, o, 16);
      float inv = 1.0f / s;
      int w = mt * 16 + j + hi * 8;
      for (int nt = 0; nt < 8; ++nt)
        probs[w * 136 + nt * 16 + lr] = f2bf(sc[mt][nt][j] * inv);
    }

  // stage v transposed in LDS: vT[d][x] so attn@v B-frags are contiguous
  for (int r = 0; r < 4; ++r) {
    int x = r * 32 + lane;
    int key = kstart + x;
    AFrag t0, t1;
    if (key >= 0 && key < NATOM) {
      const unsigned short* p = Vb + (size_t)(b * NATOM + key) * CATOM + h * DH;
      loadB(t0, p); loadB(t1, p + 16);
    } else {
      for (int i = 0; i < 8; ++i) { t0.u[i] = 0u; t1.u[i] = 0u; }
    }
    for (int d = 0; d < 16; ++d) vT[d * 136 + x] = t0.h[d];
    for (int d = 0; d < 16; ++d) vT[(d + 16) * 136 + x] = t1.h[d];
  }
  __syncthreads();

  // out = attn @ v : [32x128] x [128x32], 4 k-steps of 32
  v8f oacc[2][2];
  for (int mt = 0; mt < 2; ++mt) for (int nt = 0; nt < 2; ++nt) oacc[mt][nt] = zero8();
  for (int ks = 0; ks < 4; ++ks) {
    AFrag pa[2], vf[2];
    for (int mt = 0; mt < 2; ++mt)
      loadA(pa[mt], probs + (mt * 16 + lr) * 136 + ks * 32 + hi * 8);
    for (int nt = 0; nt < 2; ++nt)
      loadA(vf[nt], vT + (nt * 16 + lr) * 136 + ks * 32 + hi * 8);
    for (int mt = 0; mt < 2; ++mt)
      for (int nt = 0; nt < 2; ++nt)
        oacc[mt][nt] = wmma_bf16(pa[mt], vf[nt], oacc[mt][nt]);
  }
  for (int mt = 0; mt < 2; ++mt)
    for (int nt = 0; nt < 2; ++nt)
      for (int j = 0; j < 8; ++j) {
        int atom = q0 + mt * 16 + j + hi * 8;
        int c = h * DH + nt * 16 + lr;
        AO[(size_t)(b * NATOM + atom) * CATOM + c] = f2bf(oacc[mt][nt][j]);
      }
}

// ---------------------------------------------------------------------------
// out = (AO @ Wo) * mask, then LayerNorm -> bf16. Wave owns 16 rows x 256 cols
// so the LN row reduction stays inside one 16-lane half + 16 nt registers.
// ---------------------------------------------------------------------------
__global__ __launch_bounds__(256)
void k_out_ln(const unsigned short* __restrict__ AO,
              const unsigned short* __restrict__ WoT,
              const float* __restrict__ atom_mask,
              const float* __restrict__ gamma_,
              const float* __restrict__ beta_,
              unsigned short* __restrict__ XLN) {
  const int lane = threadIdx.x & 31, wave = threadIdx.x >> 5;
  const int lr = lane & 15, hi = lane >> 4;
  const int rowBase = blockIdx.x * 128 + wave * 16;
  v8f acc[16];
  for (int nt = 0; nt < 16; ++nt) acc[nt] = zero8();

  for (int k0 = 0; k0 < CATOM; k0 += 32) {
    AFrag a;
    loadA(a, AO + (size_t)(rowBase + lr) * CATOM + k0 + hi * 8);
    for (int nt = 0; nt < 16; ++nt) {
      AFrag bf;
      loadA(bf, WoT + (size_t)(nt * 16 + lr) * CATOM + k0 + hi * 8);
      acc[nt] = wmma_bf16(a, bf, acc[nt]);
    }
  }
  float g[16], be[16];
  for (int nt = 0; nt < 16; ++nt) {
    g[nt] = gamma_[nt * 16 + lr];
    be[nt] = beta_[nt * 16 + lr];
  }
  for (int j = 0; j < 8; ++j) {
    int r = rowBase + j + hi * 8;            // flattened [B*N]
    float m = atom_mask[r];
    float s1 = 0.0f, s2 = 0.0f;
    for (int nt = 0; nt < 16; ++nt) {
      float v = acc[nt][j] * m;
      acc[nt][j] = v;
      s1 += v; s2 += v * v;
    }
    for (int o = 1; o < 16; o <<= 1) {
      s1 += __shfl_xor(s1, o, 16);
      s2 += __shfl_xor(s2, o, 16);
    }
    float mean = s1 * (1.0f / CATOM);
    float var  = s2 * (1.0f / CATOM) - mean * mean;
    float inv  = rsqrtf(var + LN_EPS);
    for (int nt = 0; nt < 16; ++nt) {
      float y = (acc[nt][j] - mean) * inv * g[nt] + be[nt];
      XLN[(size_t)r * CATOM + nt * 16 + lr] = f2bf(y);
    }
  }
}

// ---------------------------------------------------------------------------
// atom_update = relu(XLN @ Ws) * mask  (f32 out, [M,384])
// ---------------------------------------------------------------------------
__global__ __launch_bounds__(256)
void k_update_gemm(const unsigned short* __restrict__ XLN,
                   const unsigned short* __restrict__ WsT,
                   const float* __restrict__ atom_mask,
                   float* __restrict__ AU) {
  const int lane = threadIdx.x & 31, wave = threadIdx.x >> 5;
  const int lr = lane & 15, hi = lane >> 4;
  const int wm = wave & 3, wn = wave >> 2;
  const int rowBase = blockIdx.x * 128 + wm * 32;
  const int colBase = blockIdx.y * 64 + wn * 32;
  v8f acc[2][2];
  for (int mt = 0; mt < 2; ++mt) for (int nt = 0; nt < 2; ++nt) acc[mt][nt] = zero8();

  for (int k0 = 0; k0 < CATOM; k0 += 32) {
    AFrag a[2], b[2];
    for (int mt = 0; mt < 2; ++mt)
      loadA(a[mt], XLN + (size_t)(rowBase + mt * 16 + lr) * CATOM + k0 + hi * 8);
    for (int nt = 0; nt < 2; ++nt)
      loadA(b[nt], WsT + (size_t)(colBase + nt * 16 + lr) * CATOM + k0 + hi * 8);
    for (int mt = 0; mt < 2; ++mt)
      for (int nt = 0; nt < 2; ++nt)
        acc[mt][nt] = wmma_bf16(a[mt], b[nt], acc[mt][nt]);
  }
  for (int mt = 0; mt < 2; ++mt)
    for (int nt = 0; nt < 2; ++nt)
      for (int j = 0; j < 8; ++j) {
        int r = rowBase + mt * 16 + j + hi * 8;
        int c = colBase + nt * 16 + lr;
        AU[(size_t)r * CS + c] = fmaxf(acc[mt][nt][j], 0.0f) * atom_mask[r];
      }
}

// ---------------------------------------------------------------------------
// segment mean into residues: s_new = (s + sum)/(count+1); indices contiguous
// ---------------------------------------------------------------------------
__global__ void k_scatter_mean(const float* __restrict__ AU,
                               const float* __restrict__ s,
                               const int* __restrict__ res_idx,
                               float* __restrict__ out) {
  int i = blockIdx.x * blockDim.x + threadIdx.x;
  if (i >= BATCH * NRESC * CS) return;
  int c = i % CS;
  int res = (i / CS) % NRESC;
  int b = i / (CS * NRESC);
  float sum = 0.0f, cnt = 0.0f;
  int base = res * APR;
  for (int a = 0; a < APR; ++a) {
    int n = base + a;
    if (res_idx[b * NATOM + n] == res) {
      sum += AU[(size_t)(b * NATOM + n) * CS + c];
      cnt += 1.0f;
    }
  }
  out[i] = (s[i] + sum) / (cnt + 1.0f);
}

// ---------------------------------------------------------------------------
extern "C" void kernel_launch(void* const* d_in, const int* in_sizes, int n_in,
                              void* d_out, int out_size, void* d_ws, size_t ws_size,
                              hipStream_t stream) {
  (void)in_sizes; (void)n_in; (void)out_size; (void)ws_size;
  const float* atom_embed = (const float*)d_in[0];
  const float* pair_bias  = (const float*)d_in[1];
  const float* atom_mask  = (const float*)d_in[2];
  const float* s          = (const float*)d_in[3];
  const float* Wq         = (const float*)d_in[4];
  const float* Wk         = (const float*)d_in[5];
  const float* Wv         = (const float*)d_in[6];
  const float* Wo         = (const float*)d_in[7];
  const float* ln_g       = (const float*)d_in[8];
  const float* ln_b       = (const float*)d_in[9];
  const float* Ws         = (const float*)d_in[10];
  const int*   ridx       = (const int*)d_in[11];
  float* out = (float*)d_out;

  char* ws = (char*)d_ws;
  // byte offsets (all 256B aligned); XLN aliases XBF, AU aliases Q|K|V (safe:
  // XBF dead after QKV GEMM, Q/K/V dead after attention; stream is in-order)
  const size_t SZ_X  = (size_t)MROWS * CATOM * 2;       // 10,485,760
  const size_t SZ_W  = (size_t)CATOM * CATOM * 2;       //    131,072
  const size_t SZ_WS = (size_t)CS * CATOM * 2;          //    196,608
  size_t o = 0;
  const size_t o_xbf = o; o += SZ_X;
  const size_t o_wqt = o; o += SZ_W;
  const size_t o_wkt = o; o += SZ_W;
  const size_t o_wvt = o; o += SZ_W;
  const size_t o_wot = o; o += SZ_W;
  const size_t o_wst = o; o += SZ_WS;
  const size_t o_q   = o; o += SZ_X;
  const size_t o_k   = o; o += SZ_X;
  const size_t o_v   = o; o += SZ_X;
  const size_t o_ao  = o; o += SZ_X;
  const size_t o_xln = o_xbf;   // alias
  const size_t o_au  = o_q;     // alias (3*SZ_X == MROWS*CS*4 exactly)

  unsigned short* xbf = (unsigned short*)(ws + o_xbf);
  unsigned short* wqt = (unsigned short*)(ws + o_wqt);
  unsigned short* wkt = (unsigned short*)(ws + o_wkt);
  unsigned short* wvt = (unsigned short*)(ws + o_wvt);
  unsigned short* wot = (unsigned short*)(ws + o_wot);
  unsigned short* wst = (unsigned short*)(ws + o_wst);
  unsigned short* qb  = (unsigned short*)(ws + o_q);
  unsigned short* kb  = (unsigned short*)(ws + o_k);
  unsigned short* vb  = (unsigned short*)(ws + o_v);
  unsigned short* ao  = (unsigned short*)(ws + o_ao);
  unsigned short* xln = (unsigned short*)(ws + o_xln);
  float*          au  = (float*)(ws + o_au);

  // 1) precision prep: activations -> bf16, weights -> transposed bf16
  k_cvt_bf16<<<4096, 256, 0, stream>>>(atom_embed, xbf, MROWS * CATOM);
  k_transpose_bf16<<<256, 256, 0, stream>>>(Wq, wqt, CATOM, CATOM);
  k_transpose_bf16<<<256, 256, 0, stream>>>(Wk, wkt, CATOM, CATOM);
  k_transpose_bf16<<<256, 256, 0, stream>>>(Wv, wvt, CATOM, CATOM);
  k_transpose_bf16<<<256, 256, 0, stream>>>(Wo, wot, CATOM, CATOM);
  k_transpose_bf16<<<384, 256, 0, stream>>>(Ws, wst, CATOM, CS);

  // 2) Q/K/V projections (WMMA bf16, f32 accum)
  dim3 gg(MROWS / 128, CATOM / 64);
  k_gemm_qkv<<<gg, 256, 0, stream>>>(xbf, wqt, qb);
  k_gemm_qkv<<<gg, 256, 0, stream>>>(xbf, wkt, kb);
  k_gemm_qkv<<<gg, 256, 0, stream>>>(xbf, wvt, vb);

  // 3) windowed attention; 139,264 B dynamic LDS (8 heads x (probs+vT))
  k_attention<<<dim3(KBLK, BATCH), 256, 8 * 2 * 32 * 136 * 2, stream>>>(
      qb, kb, vb, pair_bias, atom_mask, ao);

  // 4) Wo projection + mask + LayerNorm
  k_out_ln<<<MROWS / 128, 256, 0, stream>>>(ao, wot, atom_mask, ln_g, ln_b, xln);

  // 5) Ws projection + relu + mask
  k_update_gemm<<<dim3(MROWS / 128, CS / 64), 256, 0, stream>>>(xln, wst, atom_mask, au);

  // 6) segment mean into s
  k_scatter_mean<<<(BATCH * NRESC * CS + 255) / 256, 256, 0, stream>>>(au, s, ridx, out);
}
